// BDH_7971459301946
// MI455X (gfx1250) — compile-verified, matrix operand order checked by reading, not run on hardware
//
#include <hip/hip_runtime.h>
#include <math.h>
#include <stdint.h>

// ---------------- types ----------------
typedef __attribute__((ext_vector_type(16))) __bf16 v16bf;
typedef __attribute__((ext_vector_type(8)))  __bf16 v8bf;
typedef __attribute__((ext_vector_type(8)))  float  v8f;
typedef __attribute__((ext_vector_type(4)))  unsigned int u32x4;
typedef __attribute__((ext_vector_type(4)))  int i32x4;
typedef __attribute__((ext_vector_type(8)))  int i32x8;

// Problem constants (B=1)
#define TT 1024
#define DD 256
#define NN 8192
#define NHH 4
#define VV 256

// ---------------- toolchain discrimination for TDM builtin arity ----------------
#if defined(__has_include)
#  if __has_include(<hip/amd_detail/amd_gfx1250_TDM.h>)
#    define TDM_6ARG 1
#  endif
#endif
#if defined(__has_builtin)
#  if __has_builtin(__builtin_amdgcn_tensor_load_to_lds) && __has_builtin(__builtin_amdgcn_s_wait_tensorcnt)
#    define HAVE_TDM 1
#  endif
#endif

// ---------------- WMMA fragment load ----------------
// 16-bit A-matrix 16x32 layout (ISA 7.12.2): lane = {l15 = row (A) / col (Bt), hf = lane>>4}
// element e (0..15): k = (e<8 ? e : e+8) + 8*hf  -> two contiguous 8-element (16B) runs.
__device__ __forceinline__ v16bf frag_ld(const __bf16* p) {
  union { v16bf v; v8bf h[2]; } u;
  u.h[0] = *(const v8bf*)(p);        // k = k0 + 8*hf .. +7
  u.h[1] = *(const v8bf*)(p + 16);   // k = k0 + 16 + 8*hf .. +7
  return u.v;
}

// One wave computes a 16(M) x 64(N) tile of C = A * Bt^T.
// A  : row-major bf16 (rows M, cols K);  Bt : row-major bf16 (rows N, cols K)
__device__ __forceinline__ void wave_gemm(const __bf16* __restrict__ A, int lda, int m0,
                                          const __bf16* __restrict__ Bt, int ldb, int n0,
                                          int K, v8f acc[4]) {
  const int lane = threadIdx.x & 31;
  const int l15 = lane & 15;
  const int hf  = lane >> 4;
  const __bf16* ap  = A  + (size_t)(m0 + l15) * lda + 8 * hf;
  const __bf16* bp0 = Bt + (size_t)(n0 + l15) * ldb + 8 * hf;
  const __bf16* bp1 = bp0 + (size_t)16 * ldb;
  const __bf16* bp2 = bp0 + (size_t)32 * ldb;
  const __bf16* bp3 = bp0 + (size_t)48 * ldb;
  for (int k = 0; k < K; k += 32) {
    __builtin_prefetch(ap  + k + 512, 0, 1);   // global_prefetch_b8
    __builtin_prefetch(bp0 + k + 512, 0, 1);
    __builtin_prefetch(bp2 + k + 512, 0, 1);
    v16bf a  = frag_ld(ap  + k);
    v16bf b0 = frag_ld(bp0 + k);
    v16bf b1 = frag_ld(bp1 + k);
    v16bf b2 = frag_ld(bp2 + k);
    v16bf b3 = frag_ld(bp3 + k);
    acc[0] = __builtin_amdgcn_wmma_f32_16x16x32_bf16(false, a, false, b0, (short)0, acc[0], false, false);
    acc[1] = __builtin_amdgcn_wmma_f32_16x16x32_bf16(false, a, false, b1, (short)0, acc[1], false, false);
    acc[2] = __builtin_amdgcn_wmma_f32_16x16x32_bf16(false, a, false, b2, (short)0, acc[2], false, false);
    acc[3] = __builtin_amdgcn_wmma_f32_16x16x32_bf16(false, a, false, b3, (short)0, acc[3], false, false);
  }
}

#ifdef HAVE_TDM
// A read from an LDS-staged tile (32 rows x KC cols, row-major), B from global.
__device__ __forceinline__ void wave_gemm_ldsA(const __bf16* As, int mloc,
                                               const __bf16* __restrict__ Bt, int ldb, int n0,
                                               int kg0, int KC, v8f acc[4]) {
  const int lane = threadIdx.x & 31;
  const int l15 = lane & 15;
  const int hf  = lane >> 4;
  const __bf16* ap  = As + (mloc + l15) * KC + 8 * hf;            // ds_load path
  const __bf16* bp0 = Bt + (size_t)(n0 + l15) * ldb + kg0 + 8 * hf;
  const __bf16* bp1 = bp0 + (size_t)16 * ldb;
  const __bf16* bp2 = bp0 + (size_t)32 * ldb;
  const __bf16* bp3 = bp0 + (size_t)48 * ldb;
  for (int k = 0; k < KC; k += 32) {
    __builtin_prefetch(bp0 + k + 512, 0, 1);
    __builtin_prefetch(bp2 + k + 512, 0, 1);
    v16bf a  = frag_ld(ap  + k);
    v16bf b0 = frag_ld(bp0 + k);
    v16bf b1 = frag_ld(bp1 + k);
    v16bf b2 = frag_ld(bp2 + k);
    v16bf b3 = frag_ld(bp3 + k);
    acc[0] = __builtin_amdgcn_wmma_f32_16x16x32_bf16(false, a, false, b0, (short)0, acc[0], false, false);
    acc[1] = __builtin_amdgcn_wmma_f32_16x16x32_bf16(false, a, false, b1, (short)0, acc[1], false, false);
    acc[2] = __builtin_amdgcn_wmma_f32_16x16x32_bf16(false, a, false, b2, (short)0, acc[2], false, false);
    acc[3] = __builtin_amdgcn_wmma_f32_16x16x32_bf16(false, a, false, b3, (short)0, acc[3], false, false);
  }
}

// Issue a TDM 2-D tile load: tile (tile_d1 rows x tile_d0 elems) of a bf16 tensor
// (tensor_d1 rows x tensor_d0 elems, row stride stride0 elems) -> LDS at lds_off.
// D# layout per CDNA5 ISA sec. 8.3/8.4 (groups 2/3 zero => 2-D tensor).
__device__ __forceinline__ void tdm_load_2d(unsigned lds_off, const void* gptr,
                                            unsigned tile_d0, unsigned tile_d1,
                                            unsigned tensor_d0, unsigned tensor_d1,
                                            unsigned long long stride0) {
  unsigned long long ga = (unsigned long long)(uintptr_t)gptr;
  u32x4 g0;
  g0[0] = 1u;                                              // count=1, user descriptor
  g0[1] = lds_off;                                         // lds_addr (bytes)
  g0[2] = (unsigned)(ga & 0xFFFFFFFFu);                    // global_addr[31:0]
  g0[3] = (unsigned)((ga >> 32) & 0x01FFFFFFu) | (2u << 30); // global_addr[56:32] | type=2
  i32x8 g1;
  g1[0] = (int)(1u << 16);                                 // data_size=1 (2 bytes); mask=0
  g1[1] = (int)((tensor_d0 & 0xFFFFu) << 16);              // tensor_dim0[15:0]
  g1[2] = (int)((tensor_d0 >> 16) | ((tensor_d1 & 0xFFFFu) << 16)); // d0[31:16] | d1[15:0]
  g1[3] = (int)((tensor_d1 >> 16) | (tile_d0 << 16));      // d1[31:16] | tile_dim0
  g1[4] = (int)(tile_d1 & 0xFFFFu);                        // tile_dim1; tile_dim2=0
  g1[5] = (int)(unsigned)(stride0 & 0xFFFFFFFFull);        // dim0_stride[31:0]
  g1[6] = (int)(unsigned)((stride0 >> 32) & 0xFFFFull);    // dim0_stride[47:32]; dim1_stride lo=0
  g1[7] = 0;
  i32x4 gz = {0, 0, 0, 0};
#ifdef TDM_6ARG
  i32x8 z8 = {0, 0, 0, 0, 0, 0, 0, 0};
  __builtin_amdgcn_tensor_load_to_lds(g0, g1, gz, gz, z8, 0);
#else
  __builtin_amdgcn_tensor_load_to_lds(g0, g1, gz, gz, 0);
#endif
}
#endif  // HAVE_TDM

// ---------------- block-wide sum (wave shfl + LDS) ----------------
__device__ __forceinline__ float block_sum(float v, float* sm) {
#pragma unroll
  for (int o = 16; o > 0; o >>= 1) v += __shfl_down(v, o, 32);
  __syncthreads();
  if ((threadIdx.x & 31) == 0) sm[threadIdx.x >> 5] = v;
  __syncthreads();
  if (threadIdx.x == 0) {
    float s = 0.f;
#pragma unroll
    for (int i = 0; i < 8; ++i) s += sm[i];
    sm[0] = s;
  }
  __syncthreads();
  return sm[0];
}

// ---------------- weight convert + transpose: W[K,N] f32 -> Wt[N,K] bf16 ----------------
__global__ void __launch_bounds__(256) k_transpose_cast(const float* __restrict__ W,
                                                        __bf16* __restrict__ Wt,
                                                        int K, int Ncols) {
  size_t total = (size_t)K * Ncols;
  const float* Wz = W + (size_t)blockIdx.z * total;
  __bf16* Wtz = Wt + (size_t)blockIdx.z * total;
  size_t i = (size_t)blockIdx.x * blockDim.x + threadIdx.x;
  if (i < total) {
    int k = (int)(i / Ncols);
    int n = (int)(i % Ncols);
    Wtz[(size_t)n * K + k] = (__bf16)Wz[i];
  }
}

// ---------------- x = LN(embed[idx]) ; bf16 row-major + bf16 transposed copies ----------------
__global__ void __launch_bounds__(256) k_embed_ln(const int* __restrict__ idx,
                                                  const float* __restrict__ embed,
                                                  float* __restrict__ x,
                                                  __bf16* __restrict__ xb,
                                                  __bf16* __restrict__ xT) {
  __shared__ float sm[8];
  const int t = blockIdx.x, d = threadIdx.x;
  float v = embed[(size_t)idx[t] * DD + d];
  float s1 = block_sum(v, sm);
  float s2 = block_sum(v * v, sm);
  float m   = s1 * (1.f / DD);
  float var = s2 * (1.f / DD) - m * m;
  float y = (v - m) * rsqrtf(var + 1e-5f);
  x [(size_t)t * DD + d] = y;
  xb[(size_t)t * DD + d] = (__bf16)y;
  xT[(size_t)d * TT + t] = (__bf16)y;
}

// ---------------- x_sparse = relu(x @ enc[h]); QR = rope(x_sparse) (bf16) ----------------
__global__ void __launch_bounds__(256) k_encode(const __bf16* __restrict__ Xb,
                                                const __bf16* __restrict__ Wt,
                                                float* __restrict__ xs,
                                                __bf16* __restrict__ QR) {
  const int h = blockIdx.z;
  const int w = threadIdx.x >> 5, lane = threadIdx.x & 31;
  const int l15 = lane & 15, hf = lane >> 4;
  const int m0 = blockIdx.y * 32 + (w & 1) * 16;
  const int n0 = blockIdx.x * 256 + (w >> 1) * 64;
  v8f acc[4] = {};
  wave_gemm(Xb, DD, m0, Wt + (size_t)h * NN * DD, DD, n0, DD, acc);
  float* xsh = xs + (size_t)h * TT * NN;
  __bf16* qh = QR + (size_t)h * TT * NN;
#pragma unroll
  for (int j = 0; j < 4; ++j) {
#pragma unroll
    for (int r = 0; r < 8; ++r) {
      int t = m0 + r + 8 * hf;
      int n = n0 + 16 * j + l15;
      float f = fmaxf(acc[j][r], 0.f);
      xsh[(size_t)t * NN + n] = f;
      // RoPE: pair partner (n^1) lives in lane^1 for the same (j,r)
      float pv  = __shfl_xor(f, 1, 32);
      float rot = (n & 1) ? pv : -pv;
      // freqs[n] = 2^(-floor(n/2)/256) / (2*pi)
      float fr = exp2f((float)(n >> 1) * (-1.f / 256.f)) * 0.15915494309189535f;
      float ph = (float)t * fr;
      ph -= floorf(ph);
      float pp = 6.283185307179586f * ph;
      qh[(size_t)t * NN + n] = (__bf16)(f * __cosf(pp) + rot * __sinf(pp));
    }
  }
}

// ---------------- scores[t,s] = (QR[t]·QR[s]) * (s<t), stored bf16 ----------------
// TDM path: A tile (32 x 256 bf16) DMA'd to LDS, double-buffered; wave 0 issues
// tensor_load_to_lds + s_wait_tensorcnt; barriers publish the tile to all 8 waves.
__global__ void __launch_bounds__(256) k_scores(const __bf16* __restrict__ QR,
                                                __bf16* __restrict__ sc) {
  const int h = blockIdx.z;
  const int w = threadIdx.x >> 5, lane = threadIdx.x & 31;
  const int l15 = lane & 15, hf = lane >> 4;
  const int mblk = blockIdx.y * 32;
  const int m0 = mblk + (w & 1) * 16;
  const int n0 = blockIdx.x * 256 + (w >> 1) * 64;
  const __bf16* Q = QR + (size_t)h * TT * NN;
  v8f acc[4] = {};
#ifdef HAVE_TDM
  const int KC = 256;
  __shared__ __align__(16) __bf16 As[2][32 * 256];
  const bool live = ((int)(blockIdx.x * 256) < mblk + 31);  // block-uniform causal skip
  if (live) {
    const __bf16* arow = Q + (size_t)mblk * NN;
    if (threadIdx.x < 32)
      tdm_load_2d((unsigned)(uintptr_t)&As[0][0], arow, KC, 32, NN, TT, NN);
    const int NCH = NN / KC;  // 32 chunks
    for (int c = 0; c < NCH; ++c) {
      if (threadIdx.x < 32) {
        if (c + 1 < NCH) {
          tdm_load_2d((unsigned)(uintptr_t)&As[(c + 1) & 1][0], arow + (size_t)(c + 1) * KC,
                      KC, 32, NN, TT, NN);
          __builtin_amdgcn_s_wait_tensorcnt(1);   // chunk c complete, c+1 in flight
        } else {
          __builtin_amdgcn_s_wait_tensorcnt(0);
        }
      }
      __syncthreads();                            // LDS tile visible to all waves
      wave_gemm_ldsA(&As[c & 1][0], (w & 1) * 16, Q, NN, n0, c * KC, KC, acc);
      __syncthreads();                            // all reads done before buffer reuse
    }
  }
#else
  if (n0 < m0 + 15)  // wave-uniform skip of fully-masked causal tiles
    wave_gemm(Q, NN, m0, Q, NN, n0, NN, acc);
#endif
  __bf16* sh = sc + (size_t)h * TT * TT;
#pragma unroll
  for (int j = 0; j < 4; ++j) {
#pragma unroll
    for (int r = 0; r < 8; ++r) {
      int t = m0 + r + 8 * hf;
      int s = n0 + 16 * j + l15;
      float v = (s < t) ? acc[j][r] : 0.f;
      sh[(size_t)t * TT + s] = (__bf16)v;
    }
  }
}

// ---------------- att = scores @ x (Bt = xT) ----------------
__global__ void __launch_bounds__(256) k_att(const __bf16* __restrict__ sc,
                                             const __bf16* __restrict__ xT,
                                             float* __restrict__ att) {
  const int h = blockIdx.z;
  const int w = threadIdx.x >> 5, lane = threadIdx.x & 31;
  const int l15 = lane & 15, hf = lane >> 4;
  const int m0 = blockIdx.y * 32 + (w & 1) * 16;
  const int n0 = (w >> 1) * 64;
  v8f acc[4] = {};
  wave_gemm(sc + (size_t)h * TT * TT, TT, m0, xT, TT, n0, TT, acc);
  float* ah = att + (size_t)h * TT * DD;
#pragma unroll
  for (int j = 0; j < 4; ++j)
#pragma unroll
    for (int r = 0; r < 8; ++r) {
      int t = m0 + r + 8 * hf;
      int d = n0 + 16 * j + l15;
      ah[(size_t)t * DD + d] = acc[j][r];
    }
}

// ---------------- yKV = LN(att) -> bf16 ----------------
__global__ void __launch_bounds__(256) k_ln_att(const float* __restrict__ att,
                                                __bf16* __restrict__ yKV) {
  __shared__ float sm[8];
  const size_t row = blockIdx.x;  // h*T + t
  const int d = threadIdx.x;
  float v = att[row * DD + d];
  float s1 = block_sum(v, sm);
  float s2 = block_sum(v * v, sm);
  float m   = s1 * (1.f / DD);
  float var = s2 * (1.f / DD) - m * m;
  yKV[row * DD + d] = (__bf16)((v - m) * rsqrtf(var + 1e-5f));
}

// ---------------- y_sparse = relu(yKV @ encv[h]); xy = x_sparse * y_sparse (bf16) ----------------
__global__ void __launch_bounds__(256) k_encode_v(const __bf16* __restrict__ yKV,
                                                  const __bf16* __restrict__ Wt,
                                                  const float* __restrict__ xs,
                                                  __bf16* __restrict__ xy) {
  const int h = blockIdx.z;
  const int w = threadIdx.x >> 5, lane = threadIdx.x & 31;
  const int l15 = lane & 15, hf = lane >> 4;
  const int m0 = blockIdx.y * 32 + (w & 1) * 16;
  const int n0 = blockIdx.x * 256 + (w >> 1) * 64;
  v8f acc[4] = {};
  wave_gemm(yKV + (size_t)h * TT * DD, DD, m0, Wt + (size_t)h * NN * DD, DD, n0, DD, acc);
  const float* xsh = xs + (size_t)h * TT * NN;
#pragma unroll
  for (int j = 0; j < 4; ++j)
#pragma unroll
    for (int r = 0; r < 8; ++r) {
      int t = m0 + r + 8 * hf;
      int n = n0 + 16 * j + l15;
      float y = fmaxf(acc[j][r], 0.f);
      xy[(size_t)t * (NHH * NN) + (size_t)h * NN + n] = (__bf16)(xsh[(size_t)t * NN + n] * y);
    }
}

// ---------------- yMLP = xy @ decoder (K = 32768) ----------------
__global__ void __launch_bounds__(256) k_decoder(const __bf16* __restrict__ xy,
                                                 const __bf16* __restrict__ decT,
                                                 float* __restrict__ yMLP) {
  const int w = threadIdx.x >> 5, lane = threadIdx.x & 31;
  const int l15 = lane & 15, hf = lane >> 4;
  const int m0 = blockIdx.y * 32 + (w & 1) * 16;
  const int n0 = (w >> 1) * 64;
  v8f acc[4] = {};
  wave_gemm(xy, NHH * NN, m0, decT, NHH * NN, n0, NHH * NN, acc);
#pragma unroll
  for (int j = 0; j < 4; ++j)
#pragma unroll
    for (int r = 0; r < 8; ++r) {
      int t = m0 + r + 8 * hf;
      int d = n0 + 16 * j + l15;
      yMLP[(size_t)t * DD + d] = acc[j][r];
    }
}

// ---------------- x = LN(x + LN(yMLP)) ----------------
__global__ void __launch_bounds__(256) k_residual_ln(const float* __restrict__ yMLP,
                                                     float* __restrict__ x,
                                                     __bf16* __restrict__ xb,
                                                     __bf16* __restrict__ xT) {
  __shared__ float sm[8];
  const int t = blockIdx.x, d = threadIdx.x;
  float y = yMLP[(size_t)t * DD + d];
  float s1 = block_sum(y, sm);
  float s2 = block_sum(y * y, sm);
  float m   = s1 * (1.f / DD);
  float var = s2 * (1.f / DD) - m * m;
  float z = x[(size_t)t * DD + d] + (y - m) * rsqrtf(var + 1e-5f);
  s1 = block_sum(z, sm);
  s2 = block_sum(z * z, sm);
  m   = s1 * (1.f / DD);
  var = s2 * (1.f / DD) - m * m;
  float xn = (z - m) * rsqrtf(var + 1e-5f);
  x [(size_t)t * DD + d] = xn;
  xb[(size_t)t * DD + d] = (__bf16)xn;
  xT[(size_t)d * TT + t] = (__bf16)xn;
}

// ---------------- logits = x @ lm_head ----------------
__global__ void __launch_bounds__(256) k_logits(const __bf16* __restrict__ xb,
                                                const __bf16* __restrict__ lmT,
                                                float* __restrict__ out) {
  const int w = threadIdx.x >> 5, lane = threadIdx.x & 31;
  const int l15 = lane & 15, hf = lane >> 4;
  const int m0 = blockIdx.y * 32 + (w & 1) * 16;
  const int n0 = (w >> 1) * 64;
  v8f acc[4] = {};
  wave_gemm(xb, DD, m0, lmT, DD, n0, DD, acc);
#pragma unroll
  for (int j = 0; j < 4; ++j)
#pragma unroll
    for (int r = 0; r < 8; ++r) {
      int t = m0 + r + 8 * hf;
      int v = n0 + 16 * j + l15;
      out[(size_t)t * VV + v] = acc[j][r];
    }
}

// ---------------- host orchestration ----------------
extern "C" void kernel_launch(void* const* d_in, const int* in_sizes, int n_in,
                              void* d_out, int out_size, void* d_ws, size_t ws_size,
                              hipStream_t stream) {
  const int*   idx       = (const int*)  d_in[0];
  const float* embed     = (const float*)d_in[1];
  const float* encoder   = (const float*)d_in[2];
  const float* encoder_v = (const float*)d_in[3];
  const float* decoder   = (const float*)d_in[4];
  const float* lm_head   = (const float*)d_in[5];
  float* out = (float*)d_out;

  char* p = (char*)d_ws;
  auto take = [&](size_t bytes) -> char* {
    char* q = p;
    p += (bytes + 255) & ~(size_t)255;
    return q;
  };
  __bf16* encT  = (__bf16*)take((size_t)NHH * NN * DD * 2);   // [h][N][D]
  __bf16* encvT = (__bf16*)take((size_t)NHH * NN * DD * 2);   // [h][N][D]
  __bf16* decT  = (__bf16*)take((size_t)DD * NHH * NN * 2);   // [D][nh*N]
  __bf16* lmT   = (__bf16*)take((size_t)VV * DD * 2);         // [V][D]
  float*  x     = (float*) take((size_t)TT * DD * 4);
  __bf16* xb    = (__bf16*)take((size_t)TT * DD * 2);
  __bf16* xT    = (__bf16*)take((size_t)DD * TT * 2);
  float*  xs    = (float*) take((size_t)NHH * TT * NN * 4);   // x_sparse (f32, also output 2)
  __bf16* QR    = (__bf16*)take((size_t)NHH * TT * NN * 2);
  __bf16* sc    = (__bf16*)take((size_t)NHH * TT * TT * 2);
  float*  att   = (float*) take((size_t)NHH * TT * DD * 4);
  __bf16* yKV   = (__bf16*)take((size_t)NHH * TT * DD * 2);
  __bf16* xy    = (__bf16*)take((size_t)TT * NHH * NN * 2);
  float*  yMLP  = (float*) take((size_t)TT * DD * 4);

  dim3 b256(256);
  // Weight preprocessing (once per launch; deterministic)
  k_transpose_cast<<<dim3((DD * NN) / 256, 1, NHH), b256, 0, stream>>>(encoder,   encT,  DD, NN);
  k_transpose_cast<<<dim3((DD * NN) / 256, 1, NHH), b256, 0, stream>>>(encoder_v, encvT, DD, NN);
  k_transpose_cast<<<dim3((NHH * NN * DD) / 256, 1, 1), b256, 0, stream>>>(decoder, decT, NHH * NN, DD);
  k_transpose_cast<<<dim3((DD * VV) / 256, 1, 1), b256, 0, stream>>>(lm_head, lmT, DD, VV);

  k_embed_ln<<<dim3(TT), b256, 0, stream>>>(idx, embed, x, xb, xT);

  for (int layer = 0; layer < 6; ++layer) {
    k_encode     <<<dim3(NN / 256, TT / 32, NHH), b256, 0, stream>>>(xb, encT, xs, QR);
    k_scores     <<<dim3(TT / 256, TT / 32, NHH), b256, 0, stream>>>(QR, sc);
    k_att        <<<dim3(1,        TT / 32, NHH), b256, 0, stream>>>(sc, xT, att);
    k_ln_att     <<<dim3(NHH * TT), b256, 0, stream>>>(att, yKV);
    k_encode_v   <<<dim3(NN / 256, TT / 32, NHH), b256, 0, stream>>>(yKV, encvT, xs, xy);
    k_decoder    <<<dim3(1,        TT / 32, 1  ), b256, 0, stream>>>(xy, decT, yMLP);
    k_residual_ln<<<dim3(TT), b256, 0, stream>>>(yMLP, x, xb, xT);
  }

  k_logits<<<dim3(1, TT / 32, 1), b256, 0, stream>>>(xb, lmT, out);

  // Output 2: last-layer x_sparse [nh, T, N] flat after logits [T, V]
  hipMemcpyAsync(out + (size_t)TT * VV, xs, (size_t)NHH * TT * NN * 4,
                 hipMemcpyDeviceToDevice, stream);
}
